// Attention_54717883351553
// MI455X (gfx1250) — compile-verified, matrix operand order checked by reading, not run on hardware
//
#include <hip/hip_runtime.h>

// ---------------------------------------------------------------------------
// Problem constants (reference: B=2, N=2048, C=1024, H=16, D=64, ALPHA=0.5)
// ---------------------------------------------------------------------------
#define BB   2
#define NN   2048
#define CC   1024
#define HH   16
#define DD   64
#define ALPHA 0.5f
#define LOG2E 1.44269504088896f

typedef __bf16 bf16;
typedef __attribute__((ext_vector_type(16))) __bf16 v16bf;
typedef __attribute__((ext_vector_type(8)))  __bf16 v8bf;
typedef __attribute__((ext_vector_type(8)))  float  v8f;
typedef int v4i __attribute__((vector_size(16)));   // matches builtin param type

#if __has_builtin(__builtin_amdgcn_global_load_async_to_lds_b128)
#define HAVE_ASYNC_LDS 1
#else
#define HAVE_ASYNC_LDS 0
#endif

// ---------------------------------------------------------------------------
// WMMA helpers (CDNA5 wave32, 16x16x32 bf16 -> f32)
// A-operand: lane L holds row M=L%16; v16bf element i = K(kb+i) / K(16+kb+i),
// kb = (L<16)?0:8  => two contiguous 16-byte chunks at [kb] and [16+kb].
// B-operand mirrors A applied to B^T rows, so row-major tiles of the
// transposed B matrix load identically.
// ---------------------------------------------------------------------------
static __device__ __forceinline__ v16bf load_frag(const bf16* base, int ld, int lane) {
    const int row = lane & 15;
    const int kb  = (lane >> 4) << 3;     // 0 or 8
    const bf16* p = base + (size_t)row * (size_t)ld + kb;
    v8bf lo = *reinterpret_cast<const v8bf*>(p);
    v8bf hi = *reinterpret_cast<const v8bf*>(p + 16);
    union { v16bf v; v8bf h[2]; } u;
    u.h[0] = lo; u.h[1] = hi;
    return u.v;
}

static __device__ __forceinline__ v8f wmma_bf16(v16bf a, v16bf b, v8f c) {
    return __builtin_amdgcn_wmma_f32_16x16x32_bf16(false, a, false, b, (short)0, c, false, false);
}

static __device__ __forceinline__ float hmax16(float v) {
    v = fmaxf(v, __shfl_xor(v, 1, 32));
    v = fmaxf(v, __shfl_xor(v, 2, 32));
    v = fmaxf(v, __shfl_xor(v, 4, 32));
    v = fmaxf(v, __shfl_xor(v, 8, 32));
    return v;
}
static __device__ __forceinline__ float hsum16(float v) {
    v += __shfl_xor(v, 1, 32);
    v += __shfl_xor(v, 2, 32);
    v += __shfl_xor(v, 4, 32);
    v += __shfl_xor(v, 8, 32);
    return v;
}

// ---------------------------------------------------------------------------
// Kernel 0: fp32 -> bf16 conversion + on-device weight transposes
// ---------------------------------------------------------------------------
#define XB_ELEMS   (BB * NN * CC)          // 4194304
#define WQT_ELEMS  (3 * CC * CC)           // 3145728
#define WPT_ELEMS  (CC * CC)               // 1048576

__global__ __launch_bounds__(256) void convert_kernel(
        const float* __restrict__ x, const float* __restrict__ wqkv,
        const float* __restrict__ wproj,
        bf16* __restrict__ Xb, bf16* __restrict__ Wqkvt, bf16* __restrict__ Wprojt) {
    const int total = XB_ELEMS + WQT_ELEMS + WPT_ELEMS;
    for (int i = blockIdx.x * blockDim.x + threadIdx.x; i < total;
         i += gridDim.x * blockDim.x) {
        if (i < XB_ELEMS) {
            Xb[i] = (bf16)x[i];
        } else if (i < XB_ELEMS + WQT_ELEMS) {
            int j = i - XB_ELEMS;
            int row = j >> 10, col = j & (CC - 1);
            Wqkvt[j] = (bf16)wqkv[(size_t)col * (3 * CC) + row];
        } else {
            int j = i - XB_ELEMS - WQT_ELEMS;
            int row = j >> 10, col = j & (CC - 1);
            Wprojt[j] = (bf16)wproj[(size_t)col * CC + row];
        }
    }
}

// ---------------------------------------------------------------------------
// GEMM core: one wave computes 32(M) x 64(N).  K unrolled by 64 with true
// ping-pong register buffers (no cur<=next copies).
// ---------------------------------------------------------------------------
static __device__ __forceinline__ void load_ab(
        const bf16* __restrict__ A, int lda,
        const bf16* __restrict__ Bt, int ldb,
        int k, int lane, v16bf& a0, v16bf& a1, v16bf b[4]) {
    a0 = load_frag(A + k, lda, lane);
    a1 = load_frag(A + (size_t)16 * lda + k, lda, lane);
#pragma unroll
    for (int nt = 0; nt < 4; ++nt)
        b[nt] = load_frag(Bt + (size_t)(nt * 16) * ldb + k, ldb, lane);
}

static __device__ __forceinline__ void mma8(
        const v16bf& a0, const v16bf& a1, const v16bf b[4],
        v8f accA[4], v8f accB[4]) {
#pragma unroll
    for (int nt = 0; nt < 4; ++nt) accA[nt] = wmma_bf16(a0, b[nt], accA[nt]);
#pragma unroll
    for (int nt = 0; nt < 4; ++nt) accB[nt] = wmma_bf16(a1, b[nt], accB[nt]);
}

static __device__ __forceinline__ void gemm_wave_32x64(
        const bf16* __restrict__ A, int lda,
        const bf16* __restrict__ Bt, int ldb,
        int lane, v8f accA[4], v8f accB[4]) {
    v16bf a0A, a1A, bA[4];          // ping
    v16bf a0B, a1B, bB[4];          // pong
    load_ab(A, lda, Bt, ldb, 0, lane, a0A, a1A, bA);
    for (int k0 = 0; k0 < CC; k0 += 64) {
        load_ab(A, lda, Bt, ldb, k0 + 32, lane, a0B, a1B, bB);
        mma8(a0A, a1A, bA, accA, accB);
        const int k2 = (k0 + 64 < CC) ? (k0 + 64) : 0;   // clamped reload
        load_ab(A, lda, Bt, ldb, k2, lane, a0A, a1A, bA);
        mma8(a0B, a1B, bB, accA, accB);
    }
}

// ---------------------------------------------------------------------------
// Kernel 1: QKV GEMM [4096 x 1024] x [1024 x 3072]
//   scatter -> Q[B,H,N,D], K[B,H,N,D], Vt[B,H,D,N] (bf16)
// ---------------------------------------------------------------------------
__global__ __launch_bounds__(256) void gemm_qkv_kernel(
        const bf16* __restrict__ Xb, const bf16* __restrict__ Wqkvt,
        bf16* __restrict__ Q, bf16* __restrict__ K, bf16* __restrict__ Vt) {
    const int wave = threadIdx.x >> 5, lane = threadIdx.x & 31;
    const int m0 = blockIdx.x * 128 + (wave & 3) * 32;
    const int n0 = blockIdx.y * 128 + (wave >> 2) * 64;

    v8f accA[4] = {}, accB[4] = {};
    gemm_wave_32x64(Xb + (size_t)m0 * CC, CC, Wqkvt + (size_t)n0 * CC, CC,
                    lane, accA, accB);

    const int hi8 = (lane >> 4) << 3;
    const int l15 = lane & 15;
#pragma unroll
    for (int half = 0; half < 2; ++half) {
        const v8f* acc = half ? accB : accA;
        const int mbase = m0 + half * 16;
#pragma unroll
        for (int nt = 0; nt < 4; ++nt) {
            int j  = n0 + nt * 16 + l15;       // column in [0, 3C)
            int s  = j >> 10;                  // 0:q 1:k 2:v
            int hd = j & (CC - 1);
            int h  = hd >> 6, d = hd & 63;
#pragma unroll
            for (int r = 0; r < 8; ++r) {
                int mg = mbase + r + hi8;      // row in [0, B*N)
                int b = mg >> 11, n = mg & (NN - 1);
                bf16 v = (bf16)acc[nt][r];
                size_t bh = (size_t)b * HH + h;
                if (s == 0)      Q [(bh * NN + n) * DD + d] = v;
                else if (s == 1) K [(bh * NN + n) * DD + d] = v;
                else             Vt[(bh * DD + d) * NN + n] = v;
            }
        }
    }
}

// ---------------------------------------------------------------------------
// Attention helpers
// ---------------------------------------------------------------------------
static __device__ __forceinline__ void load_k4(
        const bf16* __restrict__ Kp, int kv, int lane, v16bf kf[4]) {
    kf[0] = load_frag(Kp + (size_t)kv * DD,             DD, lane);
    kf[1] = load_frag(Kp + (size_t)kv * DD + 32,        DD, lane);
    kf[2] = load_frag(Kp + (size_t)(kv + 16) * DD,      DD, lane);
    kf[3] = load_frag(Kp + (size_t)(kv + 16) * DD + 32, DD, lane);
}

static __device__ __forceinline__ void qk_tiles(
        const v16bf& qf0, const v16bf& qf1, const v16bf kf[4], v8f& s0, v8f& s1) {
    s0 = wmma_bf16(qf0, kf[0], s0); s0 = wmma_bf16(qf1, kf[1], s0);
    s1 = wmma_bf16(qf0, kf[2], s1); s1 = wmma_bf16(qf1, kf[3], s1);
}

#if HAVE_ASYNC_LDS
// Async copy of one 32x64 bf16 K chunk (4 KB) global -> LDS.
// Lane copies 128 contiguous bytes as 8 x b128; tracked by ASYNCcnt.
static __device__ __forceinline__ void stage_k_async(
        const bf16* __restrict__ gsrc, bf16* lds_dst, int lane) {
    v4i* g = (v4i*)((uintptr_t)gsrc + (size_t)lane * 128);
    __attribute__((address_space(3))) v4i* l =
        (__attribute__((address_space(3))) v4i*)(uint32_t)((uintptr_t)lds_dst + (size_t)lane * 128);
#pragma unroll
    for (int j = 0; j < 8; ++j)
        __builtin_amdgcn_global_load_async_to_lds_b128(g + j, l + j, 0, 0);
}
#define WAIT_ASYNC8()  asm volatile("s_wait_asynccnt 8" ::: "memory")
#define WAIT_ASYNC0()  asm volatile("s_wait_asynccnt 0" ::: "memory")
#endif
#define WAIT_DS0()     asm volatile("s_wait_dscnt 0" ::: "memory")

// ---------------------------------------------------------------------------
// Kernel 2: fused attention (two-pass softmax + distance blend + PV)
// One wave owns 16 query rows of one (b,h); block = 8 waves = 128 rows.
// grid = B*H*(N/128) = 512 blocks.
// out = ALPHA * dm @ v + (1-ALPHA) * softmax(q k^T / sqrt(D)) @ v
//
// Async variant: K chunks are prefetched into per-wave double-buffered LDS
// with GLOBAL_LOAD_ASYNC_TO_LDS_B128 (ASYNCcnt, in-order completion), then
// consumed via ds_load fragments.  Fallback: register ping-pong buffers.
// ---------------------------------------------------------------------------
__global__ __launch_bounds__(256) void attn_kernel(
        const bf16* __restrict__ Qg, const bf16* __restrict__ Kg,
        const bf16* __restrict__ Vtg, const float* __restrict__ dm,
        bf16* __restrict__ Ob) {
    __shared__ __attribute__((aligned(16))) bf16 plds[8 * 16 * 32];   // P staging, 8 KB
#if HAVE_ASYNC_LDS
    __shared__ __attribute__((aligned(16))) bf16 kstage[8][2][32 * DD]; // 64 KB
#endif

    const int wave = threadIdx.x >> 5, lane = threadIdx.x & 31;
    const int bh = blockIdx.x >> 4;               // N/128 = 16 row-blocks
    const int rb = blockIdx.x & 15;
    const int b = bh >> 4;                        // H = 16
    const int m0 = rb * 128 + wave * 16;
    const int h = bh & 15;
    const int hi8 = (lane >> 4) << 3;
    const int l15 = lane & 15;

    const bf16* Qp = Qg  + ((size_t)bh * NN + m0) * DD;
    const bf16* Kp = Kg  + (size_t)bh * NN * DD;
    const bf16* Vp = Vtg + (size_t)bh * DD * NN;
    const float* dmp = dm + ((size_t)b * NN + m0) * NN;

    const v16bf qf0 = load_frag(Qp, DD, lane);          // d = 0..31
    const v16bf qf1 = load_frag(Qp + 32, DD, lane);     // d = 32..63
    const float scale2 = 0.125f * LOG2E;                // D^-0.5 * log2(e)

    float rmax[8], rsum[8];
#pragma unroll
    for (int r = 0; r < 8; ++r) { rmax[r] = -1e30f; rsum[r] = 0.0f; }

#if HAVE_ASYNC_LDS
    bf16* kb0 = &kstage[wave][0][0];
    bf16* kb1 = &kstage[wave][1][0];

    // ---- pass 1: row max + online row sum ----
    stage_k_async(Kp, kb0, lane);
    stage_k_async(Kp + 32 * DD, kb1, lane);
    for (int kv0 = 0; kv0 < NN; kv0 += 32) {
        WAIT_ASYNC8();                             // oldest chunk has landed
        bf16* cur = (kv0 & 32) ? kb1 : kb0;
        v16bf kf[4];
        load_k4(cur, 0, lane, kf);                 // ds_load fragments
        v8f s0 = {}, s1 = {};
        qk_tiles(qf0, qf1, kf, s0, s1);
        WAIT_DS0();                                // frag reads done -> safe refill
        const int knext = (kv0 + 64 < NN) ? (kv0 + 64) : kv0;
        stage_k_async(Kp + (size_t)knext * DD, cur, lane);
#pragma unroll
        for (int r = 0; r < 8; ++r) {
            float t0 = s0[r] * scale2, t1 = s1[r] * scale2;
            float tm = hmax16(fmaxf(t0, t1));
            float nm = fmaxf(rmax[r], tm);
            float e  = exp2f(t0 - nm) + exp2f(t1 - nm);
            e = hsum16(e);
            rsum[r] = rsum[r] * exp2f(rmax[r] - nm) + e;
            rmax[r] = nm;
        }
    }
    WAIT_ASYNC0();                                 // drain before pass-2 reuse
#else
    {   // register ping-pong fallback
        v16bf kA[4], kB[4];
        load_k4(Kp, 0, lane, kA);
        for (int kv0 = 0; kv0 < NN; kv0 += 64) {
            __builtin_prefetch(Kp + (size_t)((kv0 + 128) & (NN - 1)) * DD, 0, 0);
            load_k4(Kp, kv0 + 32, lane, kB);
#pragma unroll
            for (int half = 0; half < 2; ++half) {
                v8f s0 = {}, s1 = {};
                if (half == 0) {
                    qk_tiles(qf0, qf1, kA, s0, s1);
                } else {
                    const int k2 = (kv0 + 64 < NN) ? (kv0 + 64) : 0;
                    load_k4(Kp, k2, lane, kA);
                    qk_tiles(qf0, qf1, kB, s0, s1);
                }
#pragma unroll
                for (int r = 0; r < 8; ++r) {
                    float t0 = s0[r] * scale2, t1 = s1[r] * scale2;
                    float tm = hmax16(fmaxf(t0, t1));
                    float nm = fmaxf(rmax[r], tm);
                    float e  = exp2f(t0 - nm) + exp2f(t1 - nm);
                    e = hsum16(e);
                    rsum[r] = rsum[r] * exp2f(rmax[r] - nm) + e;
                    rmax[r] = nm;
                }
            }
        }
    }
#endif

    float rinv[8];
#pragma unroll
    for (int r = 0; r < 8; ++r) rinv[r] = (1.0f - ALPHA) / rsum[r];

    // ---- pass 2: normalize, blend alpha*dm, P @ V ----
    v8f oacc[4] = {};
    bf16* L = plds + wave * (16 * 32);

#if HAVE_ASYNC_LDS
    stage_k_async(Kp, kb0, lane);
    stage_k_async(Kp + 32 * DD, kb1, lane);
    for (int kv0 = 0; kv0 < NN; kv0 += 32) {
        // V fragments for the current chunk: issue first, consume last.
        v16bf vf[4];
#pragma unroll
        for (int dt = 0; dt < 4; ++dt)
            vf[dt] = load_frag(Vp + (size_t)(dt * 16) * NN + kv0, NN, lane);

        WAIT_ASYNC8();
        bf16* cur = (kv0 & 32) ? kb1 : kb0;
        v16bf kf[4];
        load_k4(cur, 0, lane, kf);
        v8f s0 = {}, s1 = {};
        qk_tiles(qf0, qf1, kf, s0, s1);
        WAIT_DS0();
        const int knext = (kv0 + 64 < NN) ? (kv0 + 64) : kv0;
        stage_k_async(Kp + (size_t)knext * DD, cur, lane);

#pragma unroll
        for (int r = 0; r < 8; ++r) {
            int row = r + hi8;
            const float* dr = dmp + (size_t)row * NN + kv0 + l15;
            float p0 = exp2f(s0[r] * scale2 - rmax[r]) * rinv[r] + ALPHA * dr[0];
            float p1 = exp2f(s1[r] * scale2 - rmax[r]) * rinv[r] + ALPHA * dr[16];
            L[row * 32 + l15]      = (bf16)p0;
            L[row * 32 + 16 + l15] = (bf16)p1;
        }
        WAIT_DS0();                                 // P stores -> P loads
        v16bf pf = load_frag(L, 32, lane);          // A-layout readback
#pragma unroll
        for (int dt = 0; dt < 4; ++dt)
            oacc[dt] = wmma_bf16(pf, vf[dt], oacc[dt]);
    }
    WAIT_ASYNC0();
#else
    {   // register ping-pong fallback
        v16bf kA[4], kB[4];
        load_k4(Kp, 0, lane, kA);
        for (int kv0 = 0; kv0 < NN; kv0 += 64) {
#pragma unroll
            for (int half = 0; half < 2; ++half) {
                const int kv = kv0 + half * 32;
                v16bf vf[4];
#pragma unroll
                for (int dt = 0; dt < 4; ++dt)
                    vf[dt] = load_frag(Vp + (size_t)(dt * 16) * NN + kv, NN, lane);

                v8f s0 = {}, s1 = {};
                if (half == 0) {
                    qk_tiles(qf0, qf1, kA, s0, s1);
                    load_k4(Kp, kv0 + 32, lane, kB);
                } else {
                    qk_tiles(qf0, qf1, kB, s0, s1);
                    const int k2 = (kv0 + 64 < NN) ? (kv0 + 64) : 0;
                    load_k4(Kp, k2, lane, kA);
                }

#pragma unroll
                for (int r = 0; r < 8; ++r) {
                    int row = r + hi8;
                    const float* dr = dmp + (size_t)row * NN + kv + l15;
                    float p0 = exp2f(s0[r] * scale2 - rmax[r]) * rinv[r] + ALPHA * dr[0];
                    float p1 = exp2f(s1[r] * scale2 - rmax[r]) * rinv[r] + ALPHA * dr[16];
                    L[row * 32 + l15]      = (bf16)p0;
                    L[row * 32 + 16 + l15] = (bf16)p1;
                }
                WAIT_DS0();
                v16bf pf = load_frag(L, 32, lane);
#pragma unroll
                for (int dt = 0; dt < 4; ++dt)
                    oacc[dt] = wmma_bf16(pf, vf[dt], oacc[dt]);
            }
        }
    }
#endif

    // epilogue: write [B,N,C] (head-interleaved) bf16
#pragma unroll
    for (int dt = 0; dt < 4; ++dt) {
#pragma unroll
        for (int r = 0; r < 8; ++r) {
            int mg = m0 + r + hi8;
            Ob[((size_t)b * NN + mg) * CC + h * DD + dt * 16 + l15] = (bf16)oacc[dt][r];
        }
    }
}

// ---------------------------------------------------------------------------
// Kernel 3: output projection [4096 x 1024] x [1024 x 1024] + bias -> fp32
// ---------------------------------------------------------------------------
__global__ __launch_bounds__(256) void gemm_proj_kernel(
        const bf16* __restrict__ Ob, const bf16* __restrict__ Wprojt,
        const float* __restrict__ bias, float* __restrict__ out) {
    const int wave = threadIdx.x >> 5, lane = threadIdx.x & 31;
    const int m0 = blockIdx.x * 128 + (wave & 3) * 32;
    const int n0 = blockIdx.y * 128 + (wave >> 2) * 64;

    v8f accA[4] = {}, accB[4] = {};
    gemm_wave_32x64(Ob + (size_t)m0 * CC, CC, Wprojt + (size_t)n0 * CC, CC,
                    lane, accA, accB);

    const int hi8 = (lane >> 4) << 3;
    const int l15 = lane & 15;
#pragma unroll
    for (int half = 0; half < 2; ++half) {
        const v8f* acc = half ? accB : accA;
        const int mbase = m0 + half * 16;
#pragma unroll
        for (int nt = 0; nt < 4; ++nt) {
            int j = n0 + nt * 16 + l15;
            float bj = bias[j];
#pragma unroll
            for (int r = 0; r < 8; ++r) {
                int mg = mbase + r + hi8;
                out[(size_t)mg * CC + j] = acc[nt][r] + bj;
            }
        }
    }
}

// ---------------------------------------------------------------------------
// Host-side launcher
// ---------------------------------------------------------------------------
extern "C" void kernel_launch(void* const* d_in, const int* in_sizes, int n_in,
                              void* d_out, int out_size, void* d_ws, size_t ws_size,
                              hipStream_t stream) {
    const float* x     = (const float*)d_in[0];
    const float* dm    = (const float*)d_in[1];
    const float* wqkv  = (const float*)d_in[2];
    const float* wproj = (const float*)d_in[3];
    const float* bproj = (const float*)d_in[4];
    float* out = (float*)d_out;

    bf16* Xb     = (bf16*)d_ws;
    bf16* Wqkvt  = Xb     + (size_t)XB_ELEMS;
    bf16* Wprojt = Wqkvt  + (size_t)WQT_ELEMS;
    bf16* Q      = Wprojt + (size_t)WPT_ELEMS;
    bf16* K      = Q      + (size_t)BB * HH * NN * DD;
    bf16* Vt     = K      + (size_t)BB * HH * NN * DD;
    bf16* Ob     = Vt     + (size_t)BB * HH * NN * DD;

    convert_kernel<<<2048, 256, 0, stream>>>(x, wqkv, wproj, Xb, Wqkvt, Wprojt);
    gemm_qkv_kernel<<<dim3(32, 24), 256, 0, stream>>>(Xb, Wqkvt, Q, K, Vt);
    attn_kernel<<<BB * HH * (NN / 128), 256, 0, stream>>>(Q, K, Vt, dm, Ob);
    gemm_proj_kernel<<<dim3(32, 8), 256, 0, stream>>>(Ob, Wprojt, bproj, out);
}
